// BertPooler_54348516163919
// MI455X (gfx1250) — 8-bit matrix operand order repaired and checked by reading, compiled, not run on hardware
//
#include <hip/hip_runtime.h>
#include <math.h>

typedef __attribute__((ext_vector_type(8))) int v8i;

#define HDIM 1024
#define SEQ  4096
#define BATCH 8

__global__ void wabs_partial_kernel(const float* __restrict__ W,
                                    float* __restrict__ ws) {
    __shared__ float red[256];
    const int tid = threadIdx.x;
    const int gid = blockIdx.x * 256 + tid;
    float s = 0.f;
    for (int i = gid; i < HDIM * HDIM; i += 256 * 256) s += fabsf(W[i]);
    red[tid] = s;
    __syncthreads();
    for (int off = 128; off > 0; off >>= 1) {
        if (tid < off) red[tid] += red[tid + off];
        __syncthreads();
    }
    if (tid == 0) ws[blockIdx.x] = red[0];
}

__global__ void wabs_final_kernel(float* __restrict__ ws) {
    __shared__ float red[256];
    const int tid = threadIdx.x;
    red[tid] = ws[tid];
    __syncthreads();
    for (int off = 128; off > 0; off >>= 1) {
        if (tid < off) red[tid] += red[tid + off];
        __syncthreads();
    }
    if (tid == 0) ws[256] = red[0];
}

__device__ __forceinline__ int pack_signs4(const float* __restrict__ p) {
    const float4 v = *reinterpret_cast<const float4*>(p);
    const int b0 = (int)(v.x > 0.f) - (int)(v.x < 0.f);
    const int b1 = (int)(v.y > 0.f) - (int)(v.y < 0.f);
    const int b2 = (int)(v.z > 0.f) - (int)(v.z < 0.f);
    const int b3 = (int)(v.w > 0.f) - (int)(v.w < 0.f);
    return (b0 & 0xFF) | ((b1 & 0xFF) << 8) | ((b2 & 0xFF) << 16) | ((b3 & 0xFF) << 24);
}

__global__ void __launch_bounds__(256)
sign_gemm_wmma_kernel(const float* __restrict__ hs,
                      const float* __restrict__ W,
                      const float* __restrict__ bias,
                      const float* __restrict__ alpha,
                      const float* __restrict__ ws,
                      float* __restrict__ out) {
    const int lane  = threadIdx.x & 31;
    const int wave  = threadIdx.x >> 5;
    const int ntile = blockIdx.x * 8 + wave;
    const int nbase = ntile * 16;

    const int m    = lane & 15;
    const int half = lane >> 4;

    const float* arow = hs + (size_t)(m < BATCH ? m : 0) * (size_t)SEQ * HDIM;
    const float* brow = W + (size_t)(nbase + m) * HDIM;

    v8i acc = {};

    for (int kb = 0; kb < HDIM; kb += 64) {
        __builtin_prefetch(brow + kb + 64, 0, 1);

        v8i afrag, bfrag;
#pragma unroll
        for (int v = 0; v < 8; ++v) {
            const int kA = kb + ((v >> 1) << 4) + (half << 3) + ((v & 1) << 2);
            int pa = pack_signs4(arow + kA);
            afrag[v] = (m < BATCH) ? pa : 0;

            const int kB = kb + ((v >> 1) << 4) + (half << 3) + ((v & 1) << 2);
            bfrag[v] = pack_signs4(brow + kB);
        }

        acc = __builtin_amdgcn_wmma_i32_16x16x64_iu8(
             true, afrag,  true, bfrag, acc,
             false,  false);
    }

    const float wmean = ws[256] * (1.0f / (float)(HDIM * HDIM));
    const float a     = fmaxf(alpha[0], 1e-5f);
    const float scale = a * wmean;

    if (half == 0) {
        const int n = nbase + m;
        const float bn = bias[n];
#pragma unroll
        for (int r = 0; r < 8; ++r) {
            const float val = fmaf(scale, (float)acc[r], bn);
            out[(size_t)r * HDIM + n] = tanhf(val);
        }
    }
}

extern "C" void kernel_launch(void* const* d_in, const int* in_sizes, int n_in,
                              void* d_out, int out_size, void* d_ws, size_t ws_size,
                              hipStream_t stream) {
    const float* hs    = (const float*)d_in[0];
    const float* W     = (const float*)d_in[1];
    const float* bias  = (const float*)d_in[2];
    const float* alpha = (const float*)d_in[3];
    float* out = (float*)d_out;
    float* ws  = (float*)d_ws;

    wabs_partial_kernel<<<256, 256, 0, stream>>>(W, ws);
    wabs_final_kernel<<<1, 256, 0, stream>>>(ws);
    sign_gemm_wmma_kernel<<<8, 256, 0, stream>>>(hs, W, bias, alpha, ws, out);
}
